// SupervisedTEMTransition_39642548142877
// MI455X (gfx1250) — matrix operand order repaired and checked by reading, not compiled, not run on hardware
//
#include <hip/hip_runtime.h>

// ---------------------------------------------------------------------------
// SupervisedTEMTransition for MI455X (gfx1250), wave32 + WMMA f32_16x16x32_f16
// + async global->LDS double-buffered staging (ASYNCcnt) in the classifier GEMM.
//
//   out[0 .. 16384*4096)               logits  (f32)
//   out[16384*4096 .. +16384*768)      g_next  (f32)
//
// ws layout (f16 halves):
//   A16  [16384][768]   g_next in f16 (GEMM A operand)     25.2 MB
//   W16  [4096][768]    W_cls in f16                        6.3 MB
//   D''  [sum 4n*n]     per-module [n rows][4n cols]        1.3 MB
// ---------------------------------------------------------------------------

typedef _Float16 v16h __attribute__((ext_vector_type(16)));
typedef _Float16 v8h  __attribute__((ext_vector_type(8)));
typedef _Float16 v4h  __attribute__((ext_vector_type(4)));
typedef float    v8f  __attribute__((ext_vector_type(8)));
typedef float    v4f  __attribute__((ext_vector_type(4)));
typedef unsigned int v4u __attribute__((ext_vector_type(4)));

#define BATCH   16384
#define GTOT    768
#define NSTATES 4096

// D'' offsets in halves (module-major, row-major [n][4n] per module)
#define DOFF0 0
#define DOFF1 262144
#define DOFF2 524288
#define DOFF3 589824
#define DTOT  655360

// GEMM LDS staging: 128 rows x 32 halves per tile, padded row stride of
// 40 halves (80 B) -> 16-row ds_load_b128 fragment reads hit all 64 banks.
#define TPAD   40
#define BUFB   (128 * TPAD * 2)   // bytes per buffer (10240)

#define SHUF16(lo, hi) __builtin_shufflevector(lo, hi, 0,1,2,3,4,5,6,7,8,9,10,11,12,13,14,15)

// ---------------------------------------------------------------- W -> f16
__global__ __launch_bounds__(256) void convert_w_kernel(
    const float* __restrict__ W, _Float16* __restrict__ W16) {
  int i = (blockIdx.x * 256 + threadIdx.x) * 4;           // 3,145,728 elements
  v4f v = *(const v4f*)(W + i);
  *(v4h*)(W16 + i) = __builtin_convertvector(v, v4h);
}

// ------------------------------------------- D_f[a,j,i] -> D''_f[j, a*n + i]
__global__ __launch_bounds__(256) void convert_d_kernel(
    const float* __restrict__ D0, const float* __restrict__ D1,
    const float* __restrict__ D2, const float* __restrict__ D3,
    _Float16* __restrict__ Dpp) {
  int e = blockIdx.x * 256 + threadIdx.x;                 // 655,360 elements
  const float* src;
  int rem, ns;
  if (e < DOFF1)       { src = D0; rem = e;         ns = 8; }
  else if (e < DOFF2)  { src = D1; rem = e - DOFF1; ns = 8; }
  else if (e < DOFF3)  { src = D2; rem = e - DOFF2; ns = 7; }
  else                 { src = D3; rem = e - DOFF3; ns = 7; }
  int n = 1 << ns;
  int j = rem >> (ns + 2);                                // row of D''
  int c = rem & ((n << 2) - 1);                           // col = a*n + i
  int a = c >> ns;
  int i = c & (n - 1);
  Dpp[e] = (_Float16)src[(a * n + j) * n + i];
}

// --------------------------------------------------------------- transition
// 16 batch rows per block; delta = Ahat[16,4n] @ D''[4n,n] via WMMA, with the
// one-hot-masked Ahat fragments built branchlessly from g staged in LDS.
__global__ __launch_bounds__(256) void transition_kernel(
    const int* __restrict__ state, const int* __restrict__ act,
    const float* __restrict__ emb, const _Float16* __restrict__ Dpp,
    _Float16* __restrict__ A16, float* __restrict__ gnext) {
  __shared__ __attribute__((aligned(32))) float gS[16][GTOT];  // 48 KB
  __shared__ int actS[16];

  const int b0 = blockIdx.x * 16;
  for (int q = threadIdx.x; q < 16 * (GTOT / 4); q += 256) {
    int r = q / (GTOT / 4), c4 = q - r * (GTOT / 4);
    v4f v = *(const v4f*)(emb + (size_t)state[b0 + r] * GTOT + c4 * 4);
    *(v4f*)&gS[r][c4 * 4] = v;
  }
  if (threadIdx.x < 16) actS[threadIdx.x] = act[b0 + threadIdx.x];
  __syncthreads();

  const int w    = __builtin_amdgcn_readfirstlane(threadIdx.x >> 5);
  const int lane = threadIdx.x & 31;
  const int lo   = lane & 15;          // A: M row | B/C: N col
  const int hi   = lane >> 4;

  const int am = actS[lo];             // action of this lane's A row

  for (int t = w; t < 48; t += 8) {    // 48 output 16-col tiles over 8 waves
    int ns, modOff, dOffH, j0;
    if (t < 16)      { ns = 8; modOff = 0;   dOffH = DOFF0; j0 = t << 4; }
    else if (t < 32) { ns = 8; modOff = 256; dOffH = DOFF1; j0 = (t - 16) << 4; }
    else if (t < 40) { ns = 7; modOff = 512; dOffH = DOFF2; j0 = (t - 32) << 4; }
    else             { ns = 7; modOff = 640; dOffH = DOFF3; j0 = (t - 40) << 4; }
    const int n = 1 << ns, K = n << 2, nmask = n - 1;

    v8f acc = {};
    const _Float16* Bbase = Dpp + dOffH + (size_t)(j0 + lo) * K + (hi ? 16 : 0);

    for (int kk = 0; kk < K; kk += 32) {
      const int k1 = kk + (hi ? 16 : 0);
      const int k2 = k1 + 8;
      v8f g1 = *(const v8f*)&gS[lo][modOff + (k1 & nmask)];
      v8f g2 = *(const v8f*)&gS[lo][modOff + (k2 & nmask)];
      union HU { v8h h; v4u u; } u1, u2;
      u1.h = __builtin_convertvector(g1, v8h);
      u2.h = __builtin_convertvector(g2, v8h);
      const unsigned m1 = (am == (k1 >> ns)) ? 0xFFFFFFFFu : 0u;
      const unsigned m2 = (am == (k2 >> ns)) ? 0xFFFFFFFFu : 0u;
      u1.u &= (v4u){m1, m1, m1, m1};
      u2.u &= (v4u){m2, m2, m2, m2};
      v16h afrag = SHUF16(u1.h, u2.h);
      v16h bfrag = *(const v16h*)(Bbase + kk);
      acc = __builtin_amdgcn_wmma_f32_16x16x32_f16(
          false, afrag, false, bfrag, (short)0, acc, false, false);
    }

    const int gcol = modOff + j0 + lo;            // C frag: N = lo
#pragma unroll
    for (int v = 0; v < 8; v++) {
      const int mr = v + (hi ? 8 : 0);            // C frag: M = vgpr (+8)
      float r = gS[mr][gcol] + acc[v];
      r = fminf(1.0f, fmaxf(-1.0f, r));
      const size_t o = (size_t)(b0 + mr) * GTOT + gcol;
      gnext[o] = r;
      A16[o]   = (_Float16)r;
    }
  }
}

// ------------------------------------ async global->LDS stage of one K-slice
// 128x32-half A tile + 128x32-half B tile; 4 async b128 per thread.
__device__ __forceinline__ void stage_tiles(
    const _Float16* A16, const _Float16* W16, int m_blk, int n_blk, int kk,
    unsigned ldsA, unsigned ldsB, int tid) {
#pragma unroll
  for (int rep = 0; rep < 2; rep++) {
    const int q   = tid + rep * 256;          // 0..511 16-B chunks per tile
    const int row = q >> 2;
    const int c16 = q & 3;
    const unsigned loff = (unsigned)(row * (TPAD * 2) + c16 * 16);
    unsigned lA = ldsA + loff;
    unsigned lB = ldsB + loff;
    unsigned long long gA = (unsigned long long)(size_t)
        (A16 + (size_t)(m_blk + row) * GTOT + kk + c16 * 8);
    unsigned long long gB = (unsigned long long)(size_t)
        (W16 + (size_t)(n_blk + row) * GTOT + kk + c16 * 8);
    asm volatile("global_load_async_to_lds_b128 %0, %1, off"
                 :: "v"(lA), "v"(gA) : "memory");
    asm volatile("global_load_async_to_lds_b128 %0, %1, off"
                 :: "v"(lB), "v"(gB) : "memory");
  }
}

// --------------------------------------------------- logits = A @ W^T + bias
// block tile 128(M) x 128(N); 8 waves as 4x2; 32x64 per wave (2x4 WMMA tiles).
// K streamed in 32-wide slices through double-buffered LDS via async copies.
__global__ __launch_bounds__(256) void logits_gemm_kernel(
    const _Float16* __restrict__ A16, const _Float16* __restrict__ W16,
    const float* __restrict__ bias, float* __restrict__ out) {
  __shared__ __attribute__((aligned(16))) _Float16 As[2][128][TPAD];
  __shared__ __attribute__((aligned(16))) _Float16 Bs[2][128][TPAD];

  const int tid  = threadIdx.x;
  const int w    = __builtin_amdgcn_readfirstlane(tid >> 5);
  const int lane = tid & 31;
  const int lo   = lane & 15;
  const int hi   = lane >> 4;
  const int m_blk = blockIdx.y * 128;
  const int n_blk = blockIdx.x * 128;
  const int wm = (w & 3) * 32;              // wave M origin inside block
  const int wn = (w >> 2) * 64;             // wave N origin inside block

  const unsigned ldsA0 = (unsigned)(size_t)&As[0][0][0];
  const unsigned ldsB0 = (unsigned)(size_t)&Bs[0][0][0];

  v8f acc[2][4];
#pragma unroll
  for (int tj = 0; tj < 4; tj++) {
    const float b = bias[n_blk + wn + tj * 16 + lo];   // C frag: N = lo
#pragma unroll
    for (int ti = 0; ti < 2; ti++)
      acc[ti][tj] = (v8f){b, b, b, b, b, b, b, b};
  }

  // prologue: two K-slices in flight
  stage_tiles(A16, W16, m_blk, n_blk, 0,  ldsA0,        ldsB0,        tid);
  stage_tiles(A16, W16, m_blk, n_blk, 32, ldsA0 + BUFB, ldsB0 + BUFB, tid);

  for (int ks = 0; ks < 24; ks++) {         // 768 / 32 K-slices
    const int buf = ks & 1;
    if (ks < 23) asm volatile("s_wait_asynccnt 4" ::: "memory");
    else         asm volatile("s_wait_asynccnt 0" ::: "memory");
    __syncthreads();                        // whole block's slice is in LDS

    v16h af[2], bf[4];
#pragma unroll
    for (int ti = 0; ti < 2; ti++) {
      const _Float16* ap = &As[buf][wm + ti * 16 + lo][hi * 8];
      v8h a0 = *(const v8h*)ap;
      v8h a1 = *(const v8h*)(ap + 16);
      af[ti] = SHUF16(a0, a1);
    }
#pragma unroll
    for (int tj = 0; tj < 4; tj++) {
      const _Float16* bp = &Bs[buf][wn + tj * 16 + lo][hi * 8];
      v8h b0 = *(const v8h*)bp;
      v8h b1 = *(const v8h*)(bp + 16);
      bf[tj] = SHUF16(b0, b1);
    }
#pragma unroll
    for (int ti = 0; ti < 2; ti++)
#pragma unroll
      for (int tj = 0; tj < 4; tj++)
        acc[ti][tj] = __builtin_amdgcn_wmma_f32_16x16x32_f16(
            false, af[ti], false, bf[tj], (short)0, acc[ti][tj], false, false);

    __syncthreads();                        // all waves done reading buf
    if (ks + 2 < 24)
      stage_tiles(A16, W16, m_blk, n_blk, (ks + 2) * 32,
                  ldsA0 + buf * BUFB, ldsB0 + buf * BUFB, tid);
  }

#pragma unroll
  for (int ti = 0; ti < 2; ti++)
#pragma unroll
    for (int tj = 0; tj < 4; tj++) {
      const int nn = n_blk + wn + tj * 16 + lo;
#pragma unroll
      for (int v = 0; v < 8; v++) {
        const int m = m_blk + wm + ti * 16 + v + (hi ? 8 : 0);
        out[(size_t)m * NSTATES + nn] = acc[ti][tj][v];
      }
    }
}

// ---------------------------------------------------------------------------
extern "C" void kernel_launch(void* const* d_in, const int* in_sizes, int n_in,
                              void* d_out, int out_size, void* d_ws, size_t ws_size,
                              hipStream_t stream) {
  const int*   stateI = (const int*)d_in[0];
  const int*   actI   = (const int*)d_in[1];
  const float* emb    = (const float*)d_in[2];
  const float* Wcls   = (const float*)d_in[3];
  const float* bcls   = (const float*)d_in[4];
  const float* D0     = (const float*)d_in[5];
  const float* D1     = (const float*)d_in[6];
  const float* D2     = (const float*)d_in[7];
  const float* D3     = (const float*)d_in[8];

  _Float16* A16 = (_Float16*)d_ws;                       // 12,582,912 halves
  _Float16* W16 = A16 + (size_t)BATCH * GTOT;            //  3,145,728 halves
  _Float16* Dpp = W16 + (size_t)NSTATES * GTOT;          //    655,360 halves

  float* logits = (float*)d_out;
  float* gnext  = logits + (size_t)BATCH * NSTATES;

  convert_w_kernel<<<(NSTATES * GTOT / 4) / 256, 256, 0, stream>>>(Wcls, W16);
  convert_d_kernel<<<DTOT / 256, 256, 0, stream>>>(D0, D1, D2, D3, Dpp);
  transition_kernel<<<BATCH / 16, 256, 0, stream>>>(stateI, actI, emb, Dpp, A16, gnext);
  dim3 g2(NSTATES / 128, BATCH / 128);
  logits_gemm_kernel<<<g2, 256, 0, stream>>>(A16, W16, bcls, logits);
}
